// Attention_81037442941661
// MI455X (gfx1250) — matrix operand order repaired and checked by reading, not compiled, not run on hardware
//
#include <hip/hip_runtime.h>
#include <hip/hip_bf16.h>

// ---- types --------------------------------------------------------------
typedef __bf16 bf16;
typedef __attribute__((ext_vector_type(8)))  __bf16 bf16x8;
typedef __attribute__((ext_vector_type(16))) __bf16 bf16x16;
typedef __attribute__((ext_vector_type(8)))  float  v8f;

// ---- problem constants --------------------------------------------------
#define Bsz   2048
#define Tlen  200
#define Hdim  128
#define H1dim 80
#define H2dim 40
#define NEGV  (-1e9f)
#define INV_SQRT_H 0.08838834764831845f   // 1/sqrt(128)

// padded tile geometry
#define MROWS 208          // 13 tiles of 16 (200 rows + 8 pad)
#define KA    256          // folded GEMM1 K  ([k ; q*k])
#define ASTR  264          // A row stride (+8 bf16 pad -> 528B, breaks bank conflicts)
#define B1STR 264          // B1 column stride
#define K2    96           // GEMM2 K padded 80 -> 96
#define H1STR 104          // h1 row stride (+8 pad -> 208B)
#define B2STR 104
#define N2    48           // GEMM2 N padded 40 -> 48

struct __align__(16) SMem {
    bf16  A [MROWS * ASTR];    // [row][k<128 : k_bf16 | k>=128 : (q*k)_bf16]
    bf16  B1[H1dim * B1STR];   // col-major: [n][k]  (W1b - W1c ; W1d)
    bf16  h1[MROWS * H1STR];   // relu(GEMM1 + qW80), K-pad cols 80..95 zeroed
    bf16  B2[N2 * B2STR];      // col-major: [n][k]  W2 padded
    float q_l [Hdim];
    float qW80[H1dim];         // b1 + q @ (W1a + W1c)   (row-independent term)
    float b2f [N2];
    float wf48[N2];
    float scores[MROWS];
    float red[256];
};

__device__ __forceinline__ bf16x16 ld_frag(const bf16* p0, const bf16* p1) {
    bf16x8 lo = *(const bf16x8*)p0;
    bf16x8 hi = *(const bf16x8*)p1;
    bf16x16 r;
#pragma unroll
    for (int i = 0; i < 8; ++i) { r[i] = lo[i]; r[i + 8] = hi[i]; }
    return r;
}

__global__ __launch_bounds__(256, 1)
void din_attention_kernel(const float* __restrict__ query,
                          const float* __restrict__ keys,
                          const int*   __restrict__ keys_length,
                          const float* __restrict__ W1,
                          const float* __restrict__ b1,
                          const float* __restrict__ W2,
                          const float* __restrict__ b2,
                          const float* __restrict__ Wf,
                          const float* __restrict__ bfv,
                          float* __restrict__ out) {
    __shared__ SMem sm;

    const int b    = blockIdx.x;
    const int tid  = threadIdx.x;
    const int lane = tid & 31;
    const int wid  = tid >> 5;          // 0..7
    const int half = lane >> 4;         // 0/1
    const int l16  = lane & 15;

    const float* q_g = query + b * Hdim;
    const float* k_g = keys + (size_t)b * Tlen * Hdim;
    const float  bf_scalar = bfv[0];

    // ---------------- setup: stage & convert into LDS --------------------
    if (tid < Hdim) sm.q_l[tid] = q_g[tid];
    __syncthreads();

    // A rows 0..199: k (bf16) and q*k (bf16)
    for (int idx = tid; idx < Tlen * (Hdim / 4); idx += 256) {
        int t = idx >> 5;
        int j = (idx & 31) * 4;
        const float4 kv = *(const float4*)(k_g + t * Hdim + j);
        bf16* ar = &sm.A[t * ASTR];
        float q0 = sm.q_l[j], q1 = sm.q_l[j + 1], q2 = sm.q_l[j + 2], q3 = sm.q_l[j + 3];
        ar[j]     = (bf16)kv.x;  ar[j + 1] = (bf16)kv.y;
        ar[j + 2] = (bf16)kv.z;  ar[j + 3] = (bf16)kv.w;
        ar[Hdim + j]     = (bf16)(kv.x * q0);  ar[Hdim + j + 1] = (bf16)(kv.y * q1);
        ar[Hdim + j + 2] = (bf16)(kv.z * q2);  ar[Hdim + j + 3] = (bf16)(kv.w * q3);
    }
    // A pad rows 200..207 -> zero
    for (int idx = tid; idx < 8 * KA; idx += 256) {
        int r = Tlen + (idx >> 8);
        int k = idx & 255;
        sm.A[r * ASTR + k] = (bf16)0.f;
    }
    // folded B1: n-major inner index for coalesced W1 reads
    for (int idx = tid; idx < H1dim * KA; idx += 256) {
        int n = idx % H1dim;
        int k = idx / H1dim;
        float v = (k < Hdim)
                ? (W1[(Hdim + k) * H1dim + n] - W1[(2 * Hdim + k) * H1dim + n])   // W1b - W1c
                : W1[(3 * Hdim + (k - Hdim)) * H1dim + n];                        // W1d
        sm.B1[n * B1STR + k] = (bf16)v;
    }
    // qW80 = b1 + q @ (W1a + W1c)
    if (tid < H1dim) {
        float acc = b1[tid];
        for (int j = 0; j < Hdim; ++j)
            acc += sm.q_l[j] * (W1[j * H1dim + tid] + W1[(2 * Hdim + j) * H1dim + tid]);
        sm.qW80[tid] = acc;
    }
    // B2 padded (zeros in pad region), biases, Wf
    for (int idx = tid; idx < N2 * K2; idx += 256) {
        int n = idx % N2;
        int k = idx / N2;
        float v = (n < H2dim && k < H1dim) ? W2[k * H2dim + n] : 0.f;
        sm.B2[n * B2STR + k] = (bf16)v;
    }
    if (tid < N2) {
        sm.b2f [tid] = (tid < H2dim) ? b2[tid] : 0.f;
        sm.wf48[tid] = (tid < H2dim) ? Wf[tid] : 0.f;
    }
    // h1 K-pad cols 80..95 must be zero for GEMM2
    for (int idx = tid; idx < MROWS * 16; idx += 256) {
        int r = idx >> 4;
        int c = H1dim + (idx & 15);
        sm.h1[r * H1STR + c] = (bf16)0.f;
    }
    __syncthreads();

    // ---------------- GEMM1: [208 x 256] @ [256 x 80] (bf16 WMMA) --------
    for (int tile = wid; tile < 13 * 5; tile += 8) {
        int rt = tile / 5, nt = tile % 5;
        int row  = rt * 16 + l16;
        int coln = nt * 16 + l16;
        const bf16* arow = &sm.A[row * ASTR];
        const bf16* bcol = &sm.B1[coln * B1STR];
        v8f c = {};
#pragma unroll
        for (int ks = 0; ks < 8; ++ks) {
            int k0 = ks * 32;
            bf16x16 af = ld_frag(arow + k0 + half * 8, arow + k0 + half * 8 + 16);
            bf16x16 bfr = ld_frag(bcol + k0 + half * 8, bcol + k0 + half * 8 + 16);
            c = __builtin_amdgcn_wmma_f32_16x16x32_bf16(false, af, false, bfr,
                                                        (short)0, c, false, false);
        }
        float qv = sm.qW80[coln];
#pragma unroll
        for (int v = 0; v < 8; ++v) {
            int r = rt * 16 + v + half * 8;
            sm.h1[r * H1STR + coln] = (bf16)fmaxf(c[v] + qv, 0.f);
        }
    }
    __syncthreads();

    // ---------------- GEMM2 + bias/ReLU + dot(Wf) -> scores --------------
    for (int rt = wid; rt < 13; rt += 8) {
        float sc[8] = {0.f, 0.f, 0.f, 0.f, 0.f, 0.f, 0.f, 0.f};
        const bf16* arow = &sm.h1[(rt * 16 + l16) * H1STR];
#pragma unroll
        for (int nt = 0; nt < 3; ++nt) {
            const bf16* bcol = &sm.B2[(nt * 16 + l16) * B2STR];
            v8f c = {};
#pragma unroll
            for (int ks = 0; ks < 3; ++ks) {
                int k0 = ks * 32;
                bf16x16 af = ld_frag(arow + k0 + half * 8, arow + k0 + half * 8 + 16);
                bf16x16 bfr = ld_frag(bcol + k0 + half * 8, bcol + k0 + half * 8 + 16);
                c = __builtin_amdgcn_wmma_f32_16x16x32_bf16(false, af, false, bfr,
                                                            (short)0, c, false, false);
            }
            int col = nt * 16 + l16;
            float b2v = sm.b2f[col], wfv = sm.wf48[col];
#pragma unroll
            for (int v = 0; v < 8; ++v)
                sc[v] += fmaxf(c[v] + b2v, 0.f) * wfv;
        }
        // reduce over the 16 lanes (columns) of each half
#pragma unroll
        for (int v = 0; v < 8; ++v) {
            float s = sc[v];
            s += __shfl_xor(s, 1, 32);
            s += __shfl_xor(s, 2, 32);
            s += __shfl_xor(s, 4, 32);
            s += __shfl_xor(s, 8, 32);
            if (l16 == 0)
                sm.scores[rt * 16 + v + half * 8] = s + bf_scalar;
        }
    }
    __syncthreads();

    // ---------------- masked softmax over T (fp32) -----------------------
    const int len = keys_length[b];
    float s = NEGV;
    if (tid < MROWS) {
        float raw = sm.scores[tid] * INV_SQRT_H;
        s = (tid < len) ? raw : NEGV;
        sm.scores[tid] = s;
    }
    sm.red[tid] = s;
    __syncthreads();
    for (int off = 128; off > 0; off >>= 1) {
        if (tid < off) sm.red[tid] = fmaxf(sm.red[tid], sm.red[tid + off]);
        __syncthreads();
    }
    float mx = sm.red[0];
    __syncthreads();
    float p = 0.f;
    if (tid < MROWS) p = __expf(sm.scores[tid] - mx);
    sm.red[tid] = p;
    __syncthreads();
    for (int off = 128; off > 0; off >>= 1) {
        if (tid < off) sm.red[tid] += sm.red[tid + off];
        __syncthreads();
    }
    float denom = sm.red[0];
    __syncthreads();
    if (tid < MROWS) sm.scores[tid] = p / denom;
    __syncthreads();

    // ---------------- out[b,h] = sum_t attn[t] * keys[b,t,h] (fp32) ------
    {
        int h = tid & (Hdim - 1);
        int part = tid >> 7;                     // 0 / 1 : halves of T
        const float* kp = k_g + (size_t)(part * 100) * Hdim + h;
        float acc = 0.f;
#pragma unroll 4
        for (int t = 0; t < 100; ++t)
            acc += sm.scores[part * 100 + t] * kp[t * Hdim];
        sm.red[tid] = acc;
        __syncthreads();
        if (tid < Hdim)
            out[b * Hdim + tid] = sm.red[tid] + sm.red[tid + Hdim];
    }
}

extern "C" void kernel_launch(void* const* d_in, const int* in_sizes, int n_in,
                              void* d_out, int out_size, void* d_ws, size_t ws_size,
                              hipStream_t stream) {
    const float* query       = (const float*)d_in[0];
    const float* keys        = (const float*)d_in[1];
    const int*   keys_length = (const int*)  d_in[2];
    const float* W1          = (const float*)d_in[3];
    const float* b1          = (const float*)d_in[4];
    const float* W2          = (const float*)d_in[5];
    const float* b2          = (const float*)d_in[6];
    const float* Wf          = (const float*)d_in[7];
    const float* bfv         = (const float*)d_in[8];
    float* out = (float*)d_out;

    din_attention_kernel<<<Bsz, 256, 0, stream>>>(query, keys, keys_length,
                                                  W1, b1, W2, b2, Wf, bfv, out);
}
